// Attention_89970974916773
// MI455X (gfx1250) — compile-verified, matrix operand order checked by reading, not run on hardware
//
#include <hip/hip_runtime.h>
#include <hip/hip_bf16.h>
#include <math.h>

// Problem dims (fixed by the reference)
#define N_DIM 8192
#define D_DIM 2048
#define M_DIM 4096

typedef __bf16 bf16_t;
typedef __attribute__((ext_vector_type(16))) __bf16 bf16x16;
typedef __attribute__((ext_vector_type(8)))  __bf16 bf16x8;
typedef __attribute__((ext_vector_type(8)))  float  f32x8;
typedef __attribute__((ext_vector_type(4)))  float  f32x4;
typedef __attribute__((ext_vector_type(2)))  unsigned u32x2;

// fp32 -> bf16 by truncation (TF32-style)
__device__ __forceinline__ bf16_t f2bf(float f) {
  union { float f; unsigned u; } a; a.f = f;
  unsigned short h = (unsigned short)(a.u >> 16);
  union { unsigned short s; bf16_t b; } r; r.s = h;
  return r.b;
}

// Pack two fp32 into one dword of two truncated bf16 (lo, hi)
__device__ __forceinline__ unsigned pk2(float lo, float hi) {
  return (__builtin_bit_cast(unsigned, hi) & 0xFFFF0000u) |
         (__builtin_bit_cast(unsigned, lo) >> 16);
}

union FragU { bf16x16 v; bf16x8 h[2]; };

// Load one 16x32 bf16 fragment from a bf16 LDS tile row (stride 32 bf16).
// Lane L holds row (L&15); lanes 0-15 carry K={0..7,16..23}, lanes 16-31
// K={8..15,24..31} -> two ds_load_b128 straight into the operand registers.
__device__ __forceinline__ bf16x16 load_frag(const bf16_t* rowBase, int half) {
  FragU f;
  f.h[0] = *(const bf16x8*)(rowBase + half * 8);
  f.h[1] = *(const bf16x8*)(rowBase + 16 + half * 8);
  return f.v;
}

// --- gfx1250 async global->LDS staging (ASYNCcnt-tracked DMA path) ---------
__device__ __forceinline__ unsigned lds_off(const void* p) {
  // LDS aperture: addr[31:0] is the LDS byte address
  return (unsigned)(unsigned long long)p;
}
__device__ __forceinline__ void async_ld_b128(unsigned lds_addr, const void* gptr) {
  asm volatile("global_load_async_to_lds_b128 %0, %1, off"
               :: "v"(lds_addr), "v"(gptr) : "memory");
}
__device__ __forceinline__ void wait_async0() {
  asm volatile("s_wait_asynccnt 0" ::: "memory");
}

// DMA a 128-row x 32-float fp32 panel (row stride `gstride` floats) into LDS.
__device__ __forceinline__ void issue_panel_async(float* ldsDst,
                                                  const float* gsrc,
                                                  size_t gstride, int tid) {
  for (int l = tid; l < 1024; l += 256) {  // 1024 b128 moves
    const int r = l >> 3, k4 = (l & 7) * 4;
    async_ld_b128(lds_off(ldsDst + r * 32 + k4), gsrc + (size_t)r * gstride + k4);
  }
}

// DMA a 32-row x 64-float fp32 panel of x into LDS (row stride M_DIM).
__device__ __forceinline__ void issue_xpanel_async(float* ldsDst,
                                                   const float* gsrc, int tid) {
  for (int l = tid; l < 512; l += 256) {  // 512 b128 moves
    const int r = l >> 4, j4 = (l & 15) * 4;
    async_ld_b128(lds_off(ldsDst + r * 64 + j4), gsrc + (size_t)r * M_DIM + j4);
  }
}

// One-shot LDS fp32 -> LDS packed-bf16 conversion (one convert per element).
__device__ __forceinline__ void convert_panel(const float* src, bf16_t* dst,
                                              int nElem, int tid) {
  for (int l = tid * 4; l < nElem; l += 256 * 4) {
    const f32x4 v = *(const f32x4*)(src + l);
    u32x2 p;
    p[0] = pk2(v[0], v[1]);
    p[1] = pk2(v[2], v[3]);
    *(u32x2*)(dst + l) = p;  // ds_store_b64
  }
}

// ---------------------------------------------------------------------------
// Kernel: row means of x (sum / (mask_sum + 1e-5))
// ---------------------------------------------------------------------------
__global__ __launch_bounds__(256) void row_means_k(const float* __restrict__ x,
                                                   float* __restrict__ mr) {
  __shared__ float rs[256], ms[256];
  const int tid = threadIdx.x;
  const int i = blockIdx.x;
  float sx = 0.f, sm = 0.f;
  for (int j = tid; j < M_DIM; j += 256) {
    float v = x[(size_t)i * M_DIM + j];
    sx += v;
    sm += (v > 0.f) ? 1.f : 0.f;
  }
  rs[tid] = sx; ms[tid] = sm;
  __syncthreads();
  for (int s = 128; s > 0; s >>= 1) {
    if (tid < s) { rs[tid] += rs[tid + s]; ms[tid] += ms[tid + s]; }
    __syncthreads();
  }
  if (tid == 0) mr[i] = rs[0] / (ms[0] + 1e-5f);
}

// ---------------------------------------------------------------------------
// Kernel: column means of x
// ---------------------------------------------------------------------------
__global__ __launch_bounds__(256) void col_means_k(const float* __restrict__ x,
                                                   float* __restrict__ mc) {
  const int j = blockIdx.x * 256 + threadIdx.x;
  float sx = 0.f, sm = 0.f;
  for (int i = 0; i < N_DIM; ++i) {
    float v = x[(size_t)i * M_DIM + j];
    sx += v;
    sm += (v > 0.f) ? 1.f : 0.f;
  }
  mc[j] = sx / (sm + 1e-5f);
}

// ---------------------------------------------------------------------------
// Kernel: scalar means of means_rows (block 0) and means_cols (block 1)
// ---------------------------------------------------------------------------
__global__ __launch_bounds__(256) void mean_scalars_k(const float* __restrict__ mr,
                                                      const float* __restrict__ mc,
                                                      float* __restrict__ sc) {
  __shared__ float red[256];
  const int tid = threadIdx.x;
  const float* p = (blockIdx.x == 0) ? mr : mc;
  const int n = (blockIdx.x == 0) ? N_DIM : M_DIM;
  float s = 0.f;
  for (int j = tid; j < n; j += 256) s += p[j];
  red[tid] = s;
  __syncthreads();
  for (int t = 128; t > 0; t >>= 1) {
    if (tid < t) red[tid] += red[tid + t];
    __syncthreads();
  }
  if (tid == 0) sc[blockIdx.x] = red[0] / (float)n;
}

// ---------------------------------------------------------------------------
// Kernel: sims = A @ B^T (bf16 WMMA, f32 acc), diagonal zeroed.
// Block tile 128x128, KC=32. Pipeline per chunk:
//   DMA(k+1) into fp32 buf^1   ||   convert fp32 buf -> bf16 tiles ; WMMA(k)
// Worker threads issue no global loads in the hot loop.
// ---------------------------------------------------------------------------
__global__ __launch_bounds__(256) void sims_gemm_k(const float* __restrict__ A,
                                                   const float* __restrict__ B,
                                                   float* __restrict__ S) {
  __shared__ __align__(16) float  Af[2][128 * 32];  // 2 x 16KB DMA targets
  __shared__ __align__(16) float  Bf[2][128 * 32];
  __shared__ __align__(16) bf16_t Ab[128 * 32];     // 8KB bf16 tiles
  __shared__ __align__(16) bf16_t Bb[128 * 32];

  const int tid  = threadIdx.x;
  const int lane = tid & 31;
  const int wave = tid >> 5;
  const int wr = wave & 3;   // 4 row groups * 32 rows
  const int wc = wave >> 2;  // 2 col groups * 64 cols
  const int row0 = blockIdx.y * 128;
  const int col0 = blockIdx.x * 128;

  f32x8 acc[2][4] = {};

  const int half = lane >> 4;
  const int rsel = lane & 15;

  // Prologue: DMA chunk 0 into buffer 0
  issue_panel_async(Af[0], A + (size_t)row0 * D_DIM, D_DIM, tid);
  issue_panel_async(Bf[0], B + (size_t)col0 * D_DIM, D_DIM, tid);

  for (int k0 = 0; k0 < D_DIM; k0 += 32) {
    const int buf = (k0 >> 5) & 1;
    wait_async0();      // this wave's DMAs for `buf` have landed
    __syncthreads();    // all waves' DMAs landed; prev iter done reading bf16

    if (k0 + 32 < D_DIM) {  // overlap next chunk's DMA with convert+compute
      issue_panel_async(Af[buf ^ 1], A + (size_t)row0 * D_DIM + k0 + 32, D_DIM, tid);
      issue_panel_async(Bf[buf ^ 1], B + (size_t)col0 * D_DIM + k0 + 32, D_DIM, tid);
    }

    convert_panel(Af[buf], Ab, 128 * 32, tid);
    convert_panel(Bf[buf], Bb, 128 * 32, tid);
    __syncthreads();    // bf16 tiles ready

    bf16x16 af[2], bfr[4];
#pragma unroll
    for (int i = 0; i < 2; ++i)
      af[i] = load_frag(&Ab[(wr * 32 + i * 16 + rsel) * 32], half);
#pragma unroll
    for (int j = 0; j < 4; ++j)
      bfr[j] = load_frag(&Bb[(wc * 64 + j * 16 + rsel) * 32], half);

#pragma unroll
    for (int i = 0; i < 2; ++i)
#pragma unroll
      for (int j = 0; j < 4; ++j)
        acc[i][j] = __builtin_amdgcn_wmma_f32_16x16x32_bf16(
            false, af[i], false, bfr[j], (short)0, acc[i][j], false, false);
    // next iteration's top barrier protects Ab/Bb from early overwrite
  }

  // Epilogue: C/D layout — lane L: N = L&15; VGPR v: M = v + 8*(L>>4).
  // Only the 64 diagonal blocks (of 4096) can contain diagonal elements;
  // specialize so off-diagonal blocks take a plain-store path.
  const int nsel = lane & 15;
  if (blockIdx.x == blockIdx.y) {
#pragma unroll
    for (int i = 0; i < 2; ++i)
#pragma unroll
      for (int j = 0; j < 4; ++j) {
        const int col = col0 + wc * 64 + j * 16 + nsel;
#pragma unroll
        for (int v = 0; v < 8; ++v) {
          const int row = row0 + wr * 32 + i * 16 + v + half * 8;
          S[(size_t)row * N_DIM + col] = (row == col) ? 0.0f : acc[i][j][v];
        }
      }
  } else {
#pragma unroll
    for (int i = 0; i < 2; ++i)
#pragma unroll
      for (int j = 0; j < 4; ++j) {
        const int col = col0 + wc * 64 + j * 16 + nsel;
#pragma unroll
        for (int v = 0; v < 8; ++v) {
          const int row = row0 + wr * 32 + i * 16 + v + half * 8;
          S[(size_t)row * N_DIM + col] = acc[i][j][v];
        }
      }
  }
}

// ---------------------------------------------------------------------------
// Kernel: temperature[i] = pow(rowmax(sims_i) + 0.001, gamma) * beta
// ---------------------------------------------------------------------------
__global__ __launch_bounds__(256) void row_temp_k(const float* __restrict__ S,
                                                  float* __restrict__ temp,
                                                  const float* __restrict__ beta,
                                                  const float* __restrict__ gamma) {
  __shared__ float red[256];
  const int tid = threadIdx.x;
  const int i = blockIdx.x;
  const float* row = S + (size_t)i * N_DIM;
  float m = -3.402823466e38f;
  for (int j = tid; j < N_DIM; j += 256) m = fmaxf(m, row[j]);
  red[tid] = m;
  __syncthreads();
  for (int s = 128; s > 0; s >>= 1) {
    if (tid < s) red[tid] = fmaxf(red[tid], red[tid + s]);
    __syncthreads();
  }
  if (tid == 0) temp[i] = powf(red[0] + 0.001f, gamma[0]) * beta[0];
}

// ---------------------------------------------------------------------------
// Kernel: in-place row softmax of S[i,j] / temp[j]  (per-COLUMN temperature)
// ---------------------------------------------------------------------------
__global__ __launch_bounds__(256) void row_softmax_k(float* __restrict__ S,
                                                     const float* __restrict__ temp) {
  __shared__ float srow[N_DIM];  // 32KB of the 320KB WGP LDS
  __shared__ float red[256];
  __shared__ float bcast;
  const int tid = threadIdx.x;
  const int i = blockIdx.x;
  float* row = S + (size_t)i * N_DIM;

  float m = -3.402823466e38f;
  for (int j = tid; j < N_DIM; j += 256) {
    const float s = row[j] / temp[j];
    srow[j] = s;
    m = fmaxf(m, s);
  }
  red[tid] = m;
  __syncthreads();
  for (int s = 128; s > 0; s >>= 1) {
    if (tid < s) red[tid] = fmaxf(red[tid], red[tid + s]);
    __syncthreads();
  }
  if (tid == 0) bcast = red[0];
  __syncthreads();
  const float mx = bcast;

  float sum = 0.f;
  for (int j = tid; j < N_DIM; j += 256) {
    const float e = __expf(srow[j] - mx);
    srow[j] = e;
    sum += e;
  }
  __syncthreads();
  red[tid] = sum;
  __syncthreads();
  for (int s = 128; s > 0; s >>= 1) {
    if (tid < s) red[tid] += red[tid + s];
    __syncthreads();
  }
  if (tid == 0) bcast = red[0];
  __syncthreads();
  const float inv = 1.0f / bcast;

  for (int j = tid; j < N_DIM; j += 256) row[j] = srow[j] * inv;
}

// ---------------------------------------------------------------------------
// Kernel: fused triple GEMM + full epilogue.
//   mult = P@x ; scale = P@mask ; wsum = P@(mask*mr[k])
//   out  = mult/(scale+1e-4) + c0*(mr[i] - wsum/(scale+1e-4))
//        + c1*(mc[j]-mc_mean) + c2*(mr[i]-mr_mean)
// Block tile 128x64, KC=32, 12 WMMA/chunk/wave. Both the P panel and the raw
// x panel ride the async DMA engine (double-buffered); the derive pass builds
// the three bf16 B tiles from LDS-resident x (one global-free hot loop).
// ---------------------------------------------------------------------------
__global__ __launch_bounds__(256) void fused_out_k(const float* __restrict__ P,
                                                   const float* __restrict__ x,
                                                   const float* __restrict__ mr,
                                                   const float* __restrict__ mc,
                                                   const float* __restrict__ sc,
                                                   const float* __restrict__ coef,
                                                   float* __restrict__ out) {
  __shared__ __align__(16) float  Pf[2][128 * 32];  // 2 x 16KB DMA targets
  __shared__ __align__(16) float  Xf[2][32 * 64];   // 2 x 8KB DMA targets
  __shared__ __align__(16) bf16_t Pb[128 * 32];     // 8KB bf16 A tile
  __shared__ __align__(16) bf16_t Bx[64 * 32];      // 4KB each, [j][k]
  __shared__ __align__(16) bf16_t Bm[64 * 32];
  __shared__ __align__(16) bf16_t Bw[64 * 32];

  const int tid  = threadIdx.x;
  const int lane = tid & 31;
  const int wave = tid >> 5;
  const int wr = wave & 3;   // 4 row groups * 32 rows
  const int wc = wave >> 2;  // 2 col groups * 32 cols
  const int row0 = blockIdx.y * 128;
  const int col0 = blockIdx.x * 64;

  f32x8 a_mult[2][2] = {}, a_scal[2][2] = {}, a_wsum[2][2] = {};

  const int half = lane >> 4;
  const int rsel = lane & 15;

  // Prologue: DMA chunk 0 into buffer 0
  issue_panel_async(Pf[0], P + (size_t)row0 * N_DIM, N_DIM, tid);
  issue_xpanel_async(Xf[0], x + col0, tid);

  for (int k0 = 0; k0 < N_DIM; k0 += 32) {
    const int buf = (k0 >> 5) & 1;
    wait_async0();
    __syncthreads();

    if (k0 + 32 < N_DIM) {
      issue_panel_async(Pf[buf ^ 1], P + (size_t)row0 * N_DIM + k0 + 32, N_DIM, tid);
      issue_xpanel_async(Xf[buf ^ 1], x + (size_t)(k0 + 32) * M_DIM + col0, tid);
    }

    // P fp32 -> bf16 (one convert per element)
    convert_panel(Pf[buf], Pb, 128 * 32, tid);
    // Derive the 3 B tiles from LDS-resident x, transposing to [j][k]
    for (int l = tid; l < 64 * 32; l += 256) {
      const int kk = l >> 6, j = l & 63;
      const float v  = Xf[buf][kk * 64 + j];
      const float mk = (v > 0.f) ? 1.f : 0.f;
      const float w  = mk * mr[k0 + kk];
      Bx[j * 32 + kk] = f2bf(v);
      Bm[j * 32 + kk] = f2bf(mk);
      Bw[j * 32 + kk] = f2bf(w);
    }
    __syncthreads();

    bf16x16 af[2];
#pragma unroll
    for (int i = 0; i < 2; ++i)
      af[i] = load_frag(&Pb[(wr * 32 + i * 16 + rsel) * 32], half);

#pragma unroll
    for (int c = 0; c < 2; ++c) {
      const int jb = (wc * 32 + c * 16 + rsel) * 32;
      const bf16x16 bx = load_frag(&Bx[jb], half);
      const bf16x16 bm = load_frag(&Bm[jb], half);
      const bf16x16 bw = load_frag(&Bw[jb], half);
#pragma unroll
      for (int i = 0; i < 2; ++i) {
        a_mult[i][c] = __builtin_amdgcn_wmma_f32_16x16x32_bf16(
            false, af[i], false, bx, (short)0, a_mult[i][c], false, false);
        a_scal[i][c] = __builtin_amdgcn_wmma_f32_16x16x32_bf16(
            false, af[i], false, bm, (short)0, a_scal[i][c], false, false);
        a_wsum[i][c] = __builtin_amdgcn_wmma_f32_16x16x32_bf16(
            false, af[i], false, bw, (short)0, a_wsum[i][c], false, false);
      }
    }
  }

  // Epilogue
  const float c0 = coef[0], c1 = coef[1], c2 = coef[2];
  const float mrm = sc[0], mcm = sc[1];
  const int nsel = lane & 15;
#pragma unroll
  for (int i = 0; i < 2; ++i)
#pragma unroll
    for (int c = 0; c < 2; ++c) {
      const int col = col0 + wc * 32 + c * 16 + nsel;
      const float mcj = mc[col];
#pragma unroll
      for (int v = 0; v < 8; ++v) {
        const int row = row0 + wr * 32 + i * 16 + v + half * 8;
        const float mri   = mr[row];
        const float scale = a_scal[i][c][v] + 1e-4f;
        const float res   = a_mult[i][c][v] / scale;
        const float radd  = mri - a_wsum[i][c][v] / scale;
        out[(size_t)row * M_DIM + col] =
            res + c0 * radd + c1 * (mcj - mcm) + c2 * (mri - mrm);
      }
    }
}

// ---------------------------------------------------------------------------
// Launch
// ---------------------------------------------------------------------------
extern "C" void kernel_launch(void* const* d_in, const int* in_sizes, int n_in,
                              void* d_out, int out_size, void* d_ws, size_t ws_size,
                              hipStream_t stream) {
  const float* A     = (const float*)d_in[0];  // tr_onehot      [N, D]
  const float* B     = (const float*)d_in[1];  // tr_onehot_bool [N, D]
  const float* x     = (const float*)d_in[2];  // x              [N, M]
  const float* beta  = (const float*)d_in[3];  // [1]
  const float* gamma = (const float*)d_in[4];  // [1]
  const float* coef  = (const float*)d_in[5];  // [3]
  float* out = (float*)d_out;                  // [N, M]

  // Workspace layout (floats): P[N*N] | temp[N] | mr[N] | mc[M] | sc[2]
  float* P    = (float*)d_ws;
  float* temp = P + (size_t)N_DIM * N_DIM;
  float* mr   = temp + N_DIM;
  float* mc   = mr + N_DIM;
  float* sc   = mc + M_DIM;

  row_means_k   <<<N_DIM, 256, 0, stream>>>(x, mr);
  col_means_k   <<<M_DIM / 256, 256, 0, stream>>>(x, mc);
  mean_scalars_k<<<2, 256, 0, stream>>>(mr, mc, sc);

  sims_gemm_k   <<<dim3(N_DIM / 128, N_DIM / 128), 256, 0, stream>>>(A, B, P);
  row_temp_k    <<<N_DIM, 256, 0, stream>>>(P, temp, beta, gamma);
  row_softmax_k <<<N_DIM, 256, 0, stream>>>(P, temp);

  fused_out_k   <<<dim3(M_DIM / 64, N_DIM / 128), 256, 0, stream>>>(
      P, x, mr, mc, sc, coef, out);

  (void)in_sizes; (void)n_in; (void)out_size; (void)ws_size;
}
